// RelativeAttention_21165598835565
// MI455X (gfx1250) — matrix operand order repaired and checked by reading, not compiled, not run on hardware
//
#include <hip/hip_runtime.h>

#define DIM   768
#define HEADS 12
#define HD    64
#define NTOK  196
#define NPAD  224            // tokens padded to 7*32 for WMMA K-chunks
#define BATCH 64
#define MROWS (BATCH*NTOK)   // 12544 (== 196 * 64, exact)
#define QTILES 13            // ceil(196/16)
#define KTILES 14            // 224/16
#define QK_SCALE 0.125f      // 64^-0.5

typedef __bf16 bf16;
typedef __attribute__((ext_vector_type(16))) __bf16 v16bf;
typedef __attribute__((ext_vector_type(8)))  __bf16 v8bf;
typedef __attribute__((ext_vector_type(8)))  float  v8f;

// ---- WMMA bf16 fragment loaders (layouts per CDNA5 ISA 7.12.2) ----
// A (16xK row-major, lda elems): lane half h holds elems0..7 = K=k0+8h.., elems8..15 = K=k0+16+8h..
__device__ __forceinline__ v16bf load_a_frag(const bf16* __restrict__ base, int lda,
                                             int row, int k0, int half) {
    const bf16* p = base + row * lda + k0 + 8 * half;
    v8bf lo = *(const v8bf*)p;
    v8bf hi = *(const v8bf*)(p + 16);
    v16bf a;
#pragma unroll
    for (int i = 0; i < 8; ++i) { a[i] = lo[i]; a[8 + i] = hi[i]; }
    return a;
}
// B given as Bt (Ncols x K row-major): lane half h, col=l16, elems0..7 = K=k0+8h.., elems8..15 = K=k0+16+8h..
__device__ __forceinline__ v16bf load_b_frag(const bf16* __restrict__ base, int ldb,
                                             int col, int k0, int half) {
    const bf16* p = base + col * ldb + k0 + 8 * half;
    v8bf lo = *(const v8bf*)p;
    v8bf hi = *(const v8bf*)(p + 16);
    v16bf b;
#pragma unroll
    for (int i = 0; i < 8; ++i) { b[i] = lo[i]; b[8 + i] = hi[i]; }
    return b;
}
#define WMMA_BF16(A, B, C) \
    __builtin_amdgcn_wmma_f32_16x16x32_bf16(false, (A), false, (B), (short)0, (C), false, false)

// -------------------------------- elementwise prep --------------------------------
__global__ void f2bf_kernel(const float* __restrict__ in, bf16* __restrict__ out, int n) {
    int i = blockIdx.x * blockDim.x + threadIdx.x;
    if (i < n) out[i] = (bf16)in[i];
}

__global__ void bias_expand_kernel(const float* __restrict__ table, const int* __restrict__ ridx,
                                   float* __restrict__ bm) {
    int i = blockIdx.x * blockDim.x + threadIdx.x;
    if (i < HEADS * NTOK * NTOK) {
        int h = i / (NTOK * NTOK);
        int r = i % (NTOK * NTOK);
        bm[i] = table[ridx[r] * HEADS + h];
    }
}

// -------------------------------- QKV GEMM (64x64 macro-tile / wave) --------------------------------
// C[m, col] = x[m,:] . qkv_w[col,:] + qkv_b[col]; scatter into q (scaled), k, vT bf16 buffers
__global__ __launch_bounds__(256) void qkv_gemm_kernel(const bf16* __restrict__ xb,
                                                       const bf16* __restrict__ w,
                                                       const float* __restrict__ qb,
                                                       bf16* __restrict__ qd,
                                                       bf16* __restrict__ kd,
                                                       bf16* __restrict__ vt) {
    const int NT = (3 * DIM) / 64;               // 36 macro-cols
    int wave = threadIdx.x >> 5;
    int lane = threadIdx.x & 31;
    int half = lane >> 4, l16 = lane & 15;
    int tile = blockIdx.x * 8 + wave;
    int tn = tile % NT, tm = tile / NT;          // 64-wide macro tile coords

    v8f acc[4][4];
#pragma unroll
    for (int mi = 0; mi < 4; ++mi)
#pragma unroll
        for (int ni = 0; ni < 4; ++ni)
            acc[mi][ni] = (v8f){0.f, 0.f, 0.f, 0.f, 0.f, 0.f, 0.f, 0.f};

    for (int k0 = 0; k0 < DIM; k0 += 32) {
        v16bf a[4], b[4];
#pragma unroll
        for (int mi = 0; mi < 4; ++mi)
            a[mi] = load_a_frag(xb, DIM, (tm * 4 + mi) * 16 + l16, k0, half);
#pragma unroll
        for (int ni = 0; ni < 4; ++ni)
            b[ni] = load_b_frag(w, DIM, (tn * 4 + ni) * 16 + l16, k0, half);
#pragma unroll
        for (int mi = 0; mi < 4; ++mi)
#pragma unroll
            for (int ni = 0; ni < 4; ++ni)
                acc[mi][ni] = WMMA_BF16(a[mi], b[ni], acc[mi][ni]);
    }

#pragma unroll
    for (int ni = 0; ni < 4; ++ni) {
        int col = (tn * 4 + ni) * 16 + l16;      // 0..2303
        int sel = col / DIM;                     // 0=q 1=k 2=v
        int cc  = col % DIM;
        int h   = cc / HD, d = cc % HD;
        float bias = qb[col];
#pragma unroll
        for (int mi = 0; mi < 4; ++mi) {
#pragma unroll
            for (int j = 0; j < 8; ++j) {
                int m = (tm * 4 + mi) * 16 + j + 8 * half;  // < 12544 always
                int b_ = m / NTOK, n = m % NTOK;
                int bh = b_ * HEADS + h;
                float v = acc[mi][ni][j] + bias;
                if (sel == 0)      qd[(bh * NPAD + n) * HD + d] = (bf16)(v * QK_SCALE);
                else if (sel == 1) kd[(bh * NPAD + n) * HD + d] = (bf16)v;
                else               vt[(bh * HD + d) * NPAD + n] = (bf16)v; // transposed for AV
            }
        }
    }
}

// ------------------------- fused attention (QK^T + bias + softmax + AV) -------------------------
#define S1 228   // fp32 LDS row stride (228%64=36 -> conflict-free column scans)
#define S2 232   // bf16 LDS row stride (multiple of 8 -> aligned b128 A-frag reads)

__global__ __launch_bounds__(32) void attn_kernel(const bf16* __restrict__ q,
                                                  const bf16* __restrict__ k,
                                                  const bf16* __restrict__ vt,
                                                  const float* __restrict__ biasm,
                                                  bf16* __restrict__ xa) {
    __shared__ float sa[16 * S1];
    __shared__ bf16  sp[16 * S2];

    int lane = threadIdx.x & 31;
    int half = lane >> 4, l16 = lane & 15;
    int qt = blockIdx.x % QTILES;
    int bh = blockIdx.x / QTILES;
    int h  = bh % HEADS, b_ = bh / HEADS;

    const bf16*  qp = q  + (size_t)bh * NPAD * HD;
    const bf16*  kp = k  + (size_t)bh * NPAD * HD;
    const bf16*  vp = vt + (size_t)bh * HD * NPAD;
    const float* bp = biasm + (size_t)h * NTOK * NTOK;

    // q fragments for the whole head-dim (K=64 -> two chunks), kept resident
    v16bf a0 = load_a_frag(qp, HD, qt * 16 + l16, 0,  half);
    v16bf a1 = load_a_frag(qp, HD, qt * 16 + l16, 32, half);

    // ---- QK^T tiles + bias + key mask -> LDS (fp32) ----
    for (int kt = 0; kt < KTILES; ++kt) {
        v16bf b0 = load_b_frag(kp, HD, kt * 16 + l16, 0,  half);
        v16bf b1 = load_b_frag(kp, HD, kt * 16 + l16, 32, half);
        v8f c = {0.f, 0.f, 0.f, 0.f, 0.f, 0.f, 0.f, 0.f};
        c = WMMA_BF16(a0, b0, c);
        c = WMMA_BF16(a1, b1, c);
        int kg = kt * 16 + l16;                  // key column (same for all j)
#pragma unroll
        for (int j = 0; j < 8; ++j) {
            int r  = j + 8 * half;               // local query row 0..15
            int qg = qt * 16 + r;
            float v;
            if (kg < NTOK) {
                float bia = (qg < NTOK) ? bp[qg * NTOK + kg] : 0.f;
                v = c[j] + bia;
            } else {
                v = -1e30f;                      // padded keys -> softmax weight 0
            }
            sa[r * S1 + kt * 16 + l16] = v;
        }
    }
    __syncthreads();

    // ---- softmax: each lane owns (row = l16, half of the 224 columns) ----
    {
        int r  = l16;
        int c0 = half * 112;
        float m = -3.4e38f;
        for (int ci = 0; ci < 112; ++ci) m = fmaxf(m, sa[r * S1 + c0 + ci]);
        m = fmaxf(m, __shfl_xor(m, 16, 32));
        float s = 0.f;
        for (int ci = 0; ci < 112; ++ci) {
            float e = __expf(sa[r * S1 + c0 + ci] - m);
            sa[r * S1 + c0 + ci] = e;
            s += e;
        }
        s += __shfl_xor(s, 16, 32);
        float inv = 1.f / s;
        for (int ci = 0; ci < 112; ++ci)
            sp[r * S2 + c0 + ci] = (bf16)(sa[r * S1 + c0 + ci] * inv);
    }
    __syncthreads();

    // ---- attn . V : K=224 in 7 chunks; A-frag loaded once per chunk, 4 resident d-tiles ----
    v8f cacc[4];
#pragma unroll
    for (int dt = 0; dt < 4; ++dt)
        cacc[dt] = (v8f){0.f, 0.f, 0.f, 0.f, 0.f, 0.f, 0.f, 0.f};
#pragma unroll
    for (int kc = 0; kc < 7; ++kc) {
        const bf16* ap = sp + l16 * S2 + kc * 32 + 8 * half;
        v8bf lo = *(const v8bf*)ap;
        v8bf hi = *(const v8bf*)(ap + 16);
        v16bf a;
#pragma unroll
        for (int i = 0; i < 8; ++i) { a[i] = lo[i]; a[8 + i] = hi[i]; }
#pragma unroll
        for (int dt = 0; dt < 4; ++dt) {
            v16bf b = load_b_frag(vp, NPAD, dt * 16 + l16, kc * 32, half);
            cacc[dt] = WMMA_BF16(a, b, cacc[dt]);
        }
    }
#pragma unroll
    for (int dt = 0; dt < 4; ++dt) {
        int d = dt * 16 + l16;
#pragma unroll
        for (int j = 0; j < 8; ++j) {
            int qg = qt * 16 + j + 8 * half;
            if (qg < NTOK)
                xa[((size_t)b_ * NTOK + qg) * DIM + h * HD + d] = (bf16)cacc[dt][j];
        }
    }
}

// -------------------- projection GEMM + residual (64x64 macro-tile / wave) --------------------
__global__ __launch_bounds__(256) void proj_gemm_kernel(const bf16* __restrict__ xa,
                                                        const bf16* __restrict__ w,
                                                        const float* __restrict__ pb,
                                                        const float* __restrict__ inp,
                                                        float* __restrict__ out) {
    const int NT = DIM / 64;                     // 12 macro-cols
    int wave = threadIdx.x >> 5;
    int lane = threadIdx.x & 31;
    int half = lane >> 4, l16 = lane & 15;
    int tile = blockIdx.x * 8 + wave;
    int tn = tile % NT, tm = tile / NT;

    v8f acc[4][4];
#pragma unroll
    for (int mi = 0; mi < 4; ++mi)
#pragma unroll
        for (int ni = 0; ni < 4; ++ni)
            acc[mi][ni] = (v8f){0.f, 0.f, 0.f, 0.f, 0.f, 0.f, 0.f, 0.f};

    for (int k0 = 0; k0 < DIM; k0 += 32) {
        v16bf a[4], b[4];
#pragma unroll
        for (int mi = 0; mi < 4; ++mi)
            a[mi] = load_a_frag(xa, DIM, (tm * 4 + mi) * 16 + l16, k0, half);
#pragma unroll
        for (int ni = 0; ni < 4; ++ni)
            b[ni] = load_b_frag(w, DIM, (tn * 4 + ni) * 16 + l16, k0, half);
#pragma unroll
        for (int mi = 0; mi < 4; ++mi)
#pragma unroll
            for (int ni = 0; ni < 4; ++ni)
                acc[mi][ni] = WMMA_BF16(a[mi], b[ni], acc[mi][ni]);
    }

#pragma unroll
    for (int ni = 0; ni < 4; ++ni) {
        int col = (tn * 4 + ni) * 16 + l16;
        float bias = pb[col];
#pragma unroll
        for (int mi = 0; mi < 4; ++mi) {
#pragma unroll
            for (int j = 0; j < 8; ++j) {
                int m = (tm * 4 + mi) * 16 + j + 8 * half;
                out[(size_t)m * DIM + col] = acc[mi][ni][j] + bias + inp[(size_t)m * DIM + col];
            }
        }
    }
}

// -------------------------------- host launcher --------------------------------
extern "C" void kernel_launch(void* const* d_in, const int* in_sizes, int n_in,
                              void* d_out, int out_size, void* d_ws, size_t ws_size,
                              hipStream_t stream) {
    (void)in_sizes; (void)n_in; (void)out_size; (void)ws_size;

    const float* inp   = (const float*)d_in[0];   // [B,N,768]
    const float* qkv_w = (const float*)d_in[1];   // [2304,768]
    const float* qkv_b = (const float*)d_in[2];   // [2304]
    const float* prj_w = (const float*)d_in[3];   // [768,768]
    const float* prj_b = (const float*)d_in[4];   // [768]
    const float* btab  = (const float*)d_in[5];   // [729,12]
    const int*   ridx  = (const int*)  d_in[6];   // [196,196]
    float* out = (float*)d_out;

    char* ws = (char*)d_ws;
    const size_t SZ_XB  = (size_t)MROWS * DIM * 2;          // 19,267,584
    const size_t SZ_WQ  = (size_t)3 * DIM * DIM * 2;        //  3,538,944
    const size_t SZ_WP  = (size_t)DIM * DIM * 2;            //  1,179,648
    const size_t SZ_QKV = (size_t)BATCH * HEADS * NPAD * HD * 2; // 22,020,096 each
    const size_t SZ_BM  = (size_t)HEADS * NTOK * NTOK * 4;  //  1,843,968
    size_t off = 0;
    bf16*  xb    = (bf16*)(ws + off);  off += SZ_XB;
    bf16*  wq    = (bf16*)(ws + off);  off += SZ_WQ;
    bf16*  wp    = (bf16*)(ws + off);  off += SZ_WP;
    bf16*  qd    = (bf16*)(ws + off);  off += SZ_QKV;
    bf16*  kd    = (bf16*)(ws + off);  off += SZ_QKV;
    bf16*  vt    = (bf16*)(ws + off);  off += SZ_QKV;
    float* biasm = (float*)(ws + off); off += SZ_BM;
    bf16*  xa    = (bf16*)(ws + off);  off += SZ_XB;

    // zero padded token rows so padded K-columns contribute exact zeros
    hipMemsetAsync(qd, 0, SZ_QKV, stream);
    hipMemsetAsync(kd, 0, SZ_QKV, stream);
    hipMemsetAsync(vt, 0, SZ_QKV, stream);

    // fp32 -> bf16 conversions
    {
        int n0 = MROWS * DIM, n1 = 3 * DIM * DIM, n2 = DIM * DIM;
        f2bf_kernel<<<(n0 + 255) / 256, 256, 0, stream>>>(inp,   xb, n0);
        f2bf_kernel<<<(n1 + 255) / 256, 256, 0, stream>>>(qkv_w, wq, n1);
        f2bf_kernel<<<(n2 + 255) / 256, 256, 0, stream>>>(prj_w, wp, n2);
    }
    // per-head bias matrix
    {
        int n = HEADS * NTOK * NTOK;
        bias_expand_kernel<<<(n + 255) / 256, 256, 0, stream>>>(btab, ridx, biasm);
    }
    // QKV GEMM: 196 x 36 macro-tiles (64x64), 8 waves/block
    qkv_gemm_kernel<<<(MROWS / 64) * ((3 * DIM) / 64) / 8, 256, 0, stream>>>(xb, wq, qkv_b, qd, kd, vt);
    // fused attention: one wave per (b, h, qtile)
    attn_kernel<<<BATCH * HEADS * QTILES, 32, 0, stream>>>(qd, kd, vt, biasm, xa);
    // projection + residual: 196 x 12 macro-tiles (64x64)
    proj_gemm_kernel<<<(MROWS / 64) * (DIM / 64) / 8, 256, 0, stream>>>(xa, wp, prj_b, inp, out);
}
